// Net_46961172415327
// MI455X (gfx1250) — matrix operand order repaired and checked by reading, not compiled, not run on hardware
//
#include <hip/hip_runtime.h>
#include <hip/hip_bf16.h>

typedef __attribute__((ext_vector_type(16))) _Float16 v16h;
typedef __attribute__((ext_vector_type(8)))  float    v8f;

#define G_   256
#define NPER 256
#define EPER 4096
#define F_   128

// ---------------------------------------------------------------------------
// Convert W[k,n] (f32, 128x128) -> Wh[n,k] (f16, transposed) for WMMA B loads
// ---------------------------------------------------------------------------
__global__ __launch_bounds__(256) void wconv_kernel(const float* __restrict__ W,
                                                    _Float16* __restrict__ Wh) {
  int t = blockIdx.x * 256 + threadIdx.x;   // 0..16383
  int n = t >> 7, k = t & 127;
  Wh[n * F_ + k] = (_Float16)W[k * F_ + n];
}

// ---------------------------------------------------------------------------
// One-time f32 -> f16 conversion of the layer-1 node features
// ---------------------------------------------------------------------------
__global__ __launch_bounds__(256) void xconv_kernel(const float* __restrict__ x,
                                                    _Float16* __restrict__ xh) {
  size_t t = (size_t)blockIdx.x * 256 + threadIdx.x;
  xh[t] = (_Float16)x[t];
}

// ---------------------------------------------------------------------------
// Edge scatter-add: agg[g,dst] += x[g,src] (masked). One block per
// (graph, 64-feature chunk); f16 in, f32 LDS accumulate (ds_add_f32), f16 out.
// ---------------------------------------------------------------------------
__global__ __launch_bounds__(256) void agg_kernel(const _Float16* __restrict__ xin,
                                                  const int* __restrict__ src,
                                                  const int* __restrict__ dst,
                                                  const int* __restrict__ mask, // null => all valid
                                                  _Float16* __restrict__ agg, int Nn) {
  extern __shared__ float lds[];            // Nn * 64 floats (<= 64 KB)
  int g  = blockIdx.x >> 1;
  int f0 = (blockIdx.x & 1) * 64;
  int tid = threadIdx.x;
  for (int i = tid; i < Nn * 64; i += 256) lds[i] = 0.f;
  __syncthreads();
  int fl = tid & 63;
  for (int e = (tid >> 6); e < EPER; e += 4) {
    int valid = mask ? mask[(size_t)g * EPER + e] : 1;
    if (valid) {
      int s = src[(size_t)g * EPER + e];
      int d = dst[(size_t)g * EPER + e];
      float v = (float)xin[((size_t)g * Nn + s) * F_ + f0 + fl];
      atomicAdd(&lds[d * 64 + fl], v);
    }
  }
  __syncthreads();
  for (int i = tid; i < Nn * 64; i += 256) {
    int node = i >> 6, f = i & 63;
    agg[((size_t)g * Nn + node) * F_ + f0 + f] = (_Float16)lds[i];
  }
}

// ---------------------------------------------------------------------------
// Fused dual GEMM + bias (+ReLU):  out = act(A1@W1 + A2@W2 + b)
// A1,A2: f16 row-major [rows,128]; W1h,W2h: f16 N-major [128n,128k] (W^T).
// Block = 256 thr = 8 waves; each wave owns one 16x16 C tile, block owns a
// 16-row stripe; K swept in 4 steps of 32 via v_wmma_f32_16x16x32_f16.
// Fragment packing per CDNA5 ISA 7.12.2 reduces to contiguous-K runs:
//   A lane(m=l16, h=half): elems 0..7  <- A[row, k0+8h   .. +7]
//                          elems 8..15 <- A[row, k0+16+8h.. +7]
//   B lane(n=l16, h=half): elems 0..7  <- W^T[n, k0+8h   .. +7]
//                          elems 8..15 <- W^T[n, k0+16+8h.. +7]
// ---------------------------------------------------------------------------
__global__ __launch_bounds__(256) void gemm_dual_kernel(const _Float16* __restrict__ A1,
                                                        const _Float16* __restrict__ W1h,
                                                        const _Float16* __restrict__ A2,
                                                        const _Float16* __restrict__ W2h,
                                                        const float* __restrict__ bias,
                                                        float* __restrict__ out,
                                                        int relu) {
  int lane = threadIdx.x & 31;
  int wave = threadIdx.x >> 5;        // 0..7 -> N tile
  int half = lane >> 4;               // lane group
  int l16  = lane & 15;
  int m0   = blockIdx.x * 16;
  int row  = m0 + l16;                // A row this lane loads
  int ncol = wave * 16 + l16;         // C/D column this lane owns
  v8f c = {};
#pragma unroll
  for (int k0 = 0; k0 < F_; k0 += 32) {
    v16h a1, a2, b1, b2;
    const _Float16* pa1 = A1 + (size_t)row * F_ + k0 + 8 * half;
    const _Float16* pa2 = A2 + (size_t)row * F_ + k0 + 8 * half;
#pragma unroll
    for (int i = 0; i < 8; ++i) {
      a1[i]     = pa1[i];
      a1[8 + i] = pa1[16 + i];
      a2[i]     = pa2[i];
      a2[8 + i] = pa2[16 + i];
    }
    const _Float16* pb1 = W1h + (size_t)ncol * F_ + k0 + 8 * half;
    const _Float16* pb2 = W2h + (size_t)ncol * F_ + k0 + 8 * half;
#pragma unroll
    for (int i = 0; i < 8; ++i) { b1[i] = pb1[i]; b1[8 + i] = pb1[16 + i]; b2[i] = pb2[i]; b2[8 + i] = pb2[16 + i]; }
    c = __builtin_amdgcn_wmma_f32_16x16x32_f16(false, a1, false, b1, (short)0, c, false, false);
    c = __builtin_amdgcn_wmma_f32_16x16x32_f16(false, a2, false, b2, (short)0, c, false, false);
  }
  float bv = bias[ncol];
#pragma unroll
  for (int r = 0; r < 8; ++r) {       // C/D: VGPR r -> M = r + 8*half, N = l16
    float v = c[r] + bv;
    if (relu) v = fmaxf(v, 0.f);
    out[((size_t)m0 + r + 8 * half) * F_ + ncol] = v;
  }
}

// ---------------------------------------------------------------------------
// Top-k pooling: score = h @ (w/||w||); bitonic sort (desc score, asc idx);
// gather top K gated by tanh(score) -> f16; remap edges. One block per graph.
// ---------------------------------------------------------------------------
__global__ __launch_bounds__(256) void pool_kernel(const float* __restrict__ h,
                                                   const float* __restrict__ w,
                                                   const int* __restrict__ src_in,
                                                   const int* __restrict__ dst_in,
                                                   const int* __restrict__ mask_in, // null => all valid
                                                   _Float16* __restrict__ xk,
                                                   int* __restrict__ src_out,
                                                   int* __restrict__ dst_out,
                                                   int* __restrict__ mask_out,
                                                   int Nn, int K) {
  __shared__ float wn[F_];
  __shared__ float rtmp[F_];
  __shared__ float sc[NPER];
  __shared__ int   si[NPER];
  __shared__ int   inv[NPER];
  int g = blockIdx.x, tid = threadIdx.x;

  if (tid < F_) { float x = w[tid]; rtmp[tid] = x * x; }
  __syncthreads();
  for (int s2 = 64; s2 > 0; s2 >>= 1) {
    if (tid < s2) rtmp[tid] += rtmp[tid + s2];
    __syncthreads();
  }
  float invn = rsqrtf(rtmp[0]);
  if (tid < F_) wn[tid] = w[tid] * invn;
  __syncthreads();

  for (int i = tid; i < Nn; i += 256) {
    float s = 0.f;
    const float* hp = h + ((size_t)g * Nn + i) * F_;
    for (int f = 0; f < F_; ++f) s += hp[f] * wn[f];
    sc[i] = s; si[i] = i;
  }
  __syncthreads();

  // bitonic sort: ascending w.r.t. rank (higher score first; ties lower idx)
  for (int size = 2; size <= Nn; size <<= 1) {
    for (int stride = size >> 1; stride > 0; stride >>= 1) {
      for (int i = tid; i < Nn; i += 256) {
        int j = i ^ stride;
        if (j > i) {
          bool dirAsc = ((i & size) == 0);
          float s_i = sc[i], s_j = sc[j];
          int   i_i = si[i], i_j = si[j];
          bool jBeforeI = (s_j > s_i) || (s_j == s_i && i_j < i_i);
          if (jBeforeI == dirAsc) { sc[i] = s_j; sc[j] = s_i; si[i] = i_j; si[j] = i_i; }
        }
      }
      __syncthreads();
    }
  }

  for (int i = tid; i < Nn; i += 256) inv[i] = -1;
  __syncthreads();
  for (int j = tid; j < K; j += 256) inv[si[j]] = j;
  __syncthreads();

  for (int t = tid; t < K * F_; t += 256) {
    int j = t >> 7, f = t & 127;
    float gate = tanhf(sc[j]);
    xk[((size_t)g * K + j) * F_ + f] =
        (_Float16)(h[((size_t)g * Nn + si[j]) * F_ + f] * gate);
  }
  for (int e = tid; e < EPER; e += 256) {
    int s = src_in[(size_t)g * EPER + e];
    int d = dst_in[(size_t)g * EPER + e];
    int m = mask_in ? mask_in[(size_t)g * EPER + e] : 1;
    int ns = inv[s], nd = inv[d];
    int nm = (m && ns >= 0 && nd >= 0) ? 1 : 0;
    src_out[(size_t)g * EPER + e]  = ns < 0 ? 0 : ns;
    dst_out[(size_t)g * EPER + e]  = nd < 0 ? 0 : nd;
    mask_out[(size_t)g * EPER + e] = nm;
  }
}

// ---------------------------------------------------------------------------
// Readout: z[g, zoff+f] = max_j xk, z[g, zoff+128+f] = mean_j xk (f32 reduce)
// ---------------------------------------------------------------------------
__global__ __launch_bounds__(128) void readout_kernel(const _Float16* __restrict__ xk,
                                                      float* __restrict__ z,
                                                      int K, int zoff) {
  int g = blockIdx.x, f = threadIdx.x;
  float mx = -__builtin_inff(), sum = 0.f;
  for (int j = 0; j < K; ++j) {
    float v = (float)xk[((size_t)g * K + j) * F_ + f];
    mx = fmaxf(mx, v); sum += v;
  }
  z[(size_t)g * 768 + zoff + f]        = mx;
  z[(size_t)g * 768 + zoff + 128 + f]  = sum / (float)K;
}

// ---------------------------------------------------------------------------
// MLP head + log_softmax. One block (128 thr) per graph.
// ---------------------------------------------------------------------------
__global__ __launch_bounds__(128) void mlp_kernel(const float* __restrict__ z,
                                                  const float* __restrict__ Wl1, const float* __restrict__ bl1,
                                                  const float* __restrict__ Wl2, const float* __restrict__ bl2,
                                                  const float* __restrict__ Wl3, const float* __restrict__ bl3,
                                                  float* __restrict__ out) {
  __shared__ float z1[128], z2[64], s[10], red[2];
  int g = blockIdx.x, t = threadIdx.x;
  float acc = bl1[t];
  const float* zp = z + (size_t)g * 768;
  for (int i = 0; i < 768; ++i) acc += zp[i] * Wl1[(size_t)i * 128 + t];
  z1[t] = fmaxf(acc, 0.f);
  __syncthreads();
  if (t < 64) {
    float a2 = bl2[t];
    for (int i = 0; i < 128; ++i) a2 += z1[i] * Wl2[(size_t)i * 64 + t];
    z2[t] = fmaxf(a2, 0.f);
  }
  __syncthreads();
  if (t < 10) {
    float a3 = bl3[t];
    for (int i = 0; i < 64; ++i) a3 += z2[i] * Wl3[(size_t)i * 10 + t];
    s[t] = a3;
  }
  __syncthreads();
  if (t == 0) {
    float m = s[0];
    for (int i = 1; i < 10; ++i) m = fmaxf(m, s[i]);
    float se = 0.f;
    for (int i = 0; i < 10; ++i) se += expf(s[i] - m);
    red[0] = m; red[1] = logf(se);
  }
  __syncthreads();
  if (t < 10) out[(size_t)g * 10 + t] = s[t] - red[0] - red[1];
}

// ---------------------------------------------------------------------------
extern "C" void kernel_launch(void* const* d_in, const int* in_sizes, int n_in,
                              void* d_out, int out_size, void* d_ws, size_t ws_size,
                              hipStream_t stream) {
  const float* x       = (const float*)d_in[0];
  const int*   src     = (const int*)d_in[1];
  const int*   dst     = (const int*)d_in[2];
  const float* W_root1 = (const float*)d_in[3];
  const float* W_rel1  = (const float*)d_in[4];
  const float* b1      = (const float*)d_in[5];
  const float* wp1     = (const float*)d_in[6];
  const float* W_root2 = (const float*)d_in[7];
  const float* W_rel2  = (const float*)d_in[8];
  const float* b2      = (const float*)d_in[9];
  const float* wp2     = (const float*)d_in[10];
  const float* W_root3 = (const float*)d_in[11];
  const float* W_rel3  = (const float*)d_in[12];
  const float* b3      = (const float*)d_in[13];
  const float* wp3     = (const float*)d_in[14];
  const float* Wl1     = (const float*)d_in[15];
  const float* bl1     = (const float*)d_in[16];
  const float* Wl2     = (const float*)d_in[17];
  const float* bl2     = (const float*)d_in[18];
  const float* Wl3     = (const float*)d_in[19];
  const float* bl3     = (const float*)d_in[20];
  float* out = (float*)d_out;

  char* ws = (char*)d_ws;
  const size_t WH = (size_t)F_ * F_ * sizeof(_Float16);   // 32 KB per matrix
  _Float16* Wrel1h  = (_Float16*)(ws + 0 * WH);
  _Float16* Wroot1h = (_Float16*)(ws + 1 * WH);
  _Float16* Wrel2h  = (_Float16*)(ws + 2 * WH);
  _Float16* Wroot2h = (_Float16*)(ws + 3 * WH);
  _Float16* Wrel3h  = (_Float16*)(ws + 4 * WH);
  _Float16* Wroot3h = (_Float16*)(ws + 5 * WH);
  size_t off = 6 * WH;                                     // 196608
  _Float16* xh   = (_Float16*)(ws + off); off += (size_t)G_ * NPER * F_ * 2; // 16.8 MB
  _Float16* aggh = (_Float16*)(ws + off); off += (size_t)G_ * NPER * F_ * 2; // 16.8 MB
  float*    hbf  = (float*)(ws + off);    off += (size_t)G_ * NPER * F_ * 4; // 33.5 MB
  _Float16* xk1  = (_Float16*)(ws + off); off += (size_t)G_ * 128 * F_ * 2;  //  8.4 MB
  _Float16* xk2  = (_Float16*)(ws + off); off += (size_t)G_ * 64  * F_ * 2;  //  4.2 MB
  _Float16* xk3  = (_Float16*)(ws + off); off += (size_t)G_ * 32  * F_ * 2;  //  2.1 MB
  int* ns1 = (int*)(ws + off); off += (size_t)G_ * EPER * 4;
  int* nd1 = (int*)(ws + off); off += (size_t)G_ * EPER * 4;
  int* nm1 = (int*)(ws + off); off += (size_t)G_ * EPER * 4;
  int* ns2 = (int*)(ws + off); off += (size_t)G_ * EPER * 4;
  int* nd2 = (int*)(ws + off); off += (size_t)G_ * EPER * 4;
  int* nm2 = (int*)(ws + off); off += (size_t)G_ * EPER * 4;
  float* zbuf = (float*)(ws + off); off += (size_t)G_ * 768 * 4;
  (void)ws_size; (void)in_sizes; (void)n_in; (void)out_size;

  dim3 b256(256), b128(128);

  // one-time-per-call weight transpose+convert (reused G_ times each)
  wconv_kernel<<<64, b256, 0, stream>>>(W_rel1,  Wrel1h);
  wconv_kernel<<<64, b256, 0, stream>>>(W_root1, Wroot1h);
  wconv_kernel<<<64, b256, 0, stream>>>(W_rel2,  Wrel2h);
  wconv_kernel<<<64, b256, 0, stream>>>(W_root2, Wroot2h);
  wconv_kernel<<<64, b256, 0, stream>>>(W_rel3,  Wrel3h);
  wconv_kernel<<<64, b256, 0, stream>>>(W_root3, Wroot3h);
  xconv_kernel<<<(G_ * NPER * F_) / 256, b256, 0, stream>>>(x, xh);

  // ----- layer 1 (Nn=256, K=128) -----
  agg_kernel<<<G_ * 2, b256, NPER * 64 * 4, stream>>>(xh, src, dst, nullptr, aggh, NPER);
  gemm_dual_kernel<<<(G_ * NPER) / 16, b256, 0, stream>>>(aggh, Wrel1h, xh, Wroot1h, b1, hbf, 1);
  pool_kernel<<<G_, b256, 0, stream>>>(hbf, wp1, src, dst, nullptr, xk1, ns1, nd1, nm1, NPER, 128);
  readout_kernel<<<G_, b128, 0, stream>>>(xk1, zbuf, 128, 0);

  // ----- layer 2 (Nn=128, K=64) -----
  agg_kernel<<<G_ * 2, b256, 128 * 64 * 4, stream>>>(xk1, ns1, nd1, nm1, aggh, 128);
  gemm_dual_kernel<<<(G_ * 128) / 16, b256, 0, stream>>>(aggh, Wrel2h, xk1, Wroot2h, b2, hbf, 1);
  pool_kernel<<<G_, b256, 0, stream>>>(hbf, wp2, ns1, nd1, nm1, xk2, ns2, nd2, nm2, 128, 64);
  readout_kernel<<<G_, b128, 0, stream>>>(xk2, zbuf, 64, 256);

  // ----- layer 3 (Nn=64, K=32) ----- (layer-3 pooled edges reuse set 1; dead after)
  agg_kernel<<<G_ * 2, b256, 64 * 64 * 4, stream>>>(xk2, ns2, nd2, nm2, aggh, 64);
  gemm_dual_kernel<<<(G_ * 64) / 16, b256, 0, stream>>>(aggh, Wrel3h, xk2, Wroot3h, b3, hbf, 1);
  pool_kernel<<<G_, b256, 0, stream>>>(hbf, wp3, ns2, nd2, nm2, xk3, ns1, nd1, nm1, 64, 32);
  readout_kernel<<<G_, b128, 0, stream>>>(xk3, zbuf, 32, 512);

  // ----- MLP head + log_softmax -----
  mlp_kernel<<<G_, b128, 0, stream>>>(zbuf, Wl1, bl1, Wl2, bl2, Wl3, bl3, out);
}
